// Decoder_6863357739455
// MI455X (gfx1250) — matrix operand order repaired and checked by reading, not compiled, not run on hardware
//
#include <hip/hip_runtime.h>
#include <hip/hip_bf16.h>
#include <math.h>

#define B 32
#define T 1024
#define D 512
#define D2 1024
#define D4 2048
#define E 256
#define V 50257
#define EXTRA 200
#define VE 50457            // V + EXTRA
#define NPAD 50272          // V padded to multiple of 16 for WMMA N tiles
#define KIN 1280            // 2D + E
#define PGK 2304            // 4D + E

typedef __attribute__((ext_vector_type(16))) __bf16    v16bf;
typedef __attribute__((ext_vector_type(8)))  float     v8f;
typedef __attribute__((ext_vector_type(4)))  unsigned  uv4;
typedef __attribute__((ext_vector_type(2)))  unsigned  uv2;
typedef __attribute__((ext_vector_type(4)))  float     fv4;

union FragBF { uv4 u[2]; v16bf v; };

// Native V_TANH_F32 if the toolchain exposes it (CDNA5 TRANS op, co-executes
// with the XDL/WMMA pipe); guarded so host pass / older clang fall back to libm.
#if defined(__has_builtin)
#  if __has_builtin(__builtin_amdgcn_tanhf)
#    define FAST_TANH(x) __builtin_amdgcn_tanhf(x)
#  elif __has_builtin(__builtin_amdgcn_tanh_f32)
#    define FAST_TANH(x) __builtin_amdgcn_tanh_f32(x)
#  endif
#endif
#ifndef FAST_TANH
#  define FAST_TANH(x) tanhf(x)
#endif

__device__ __forceinline__ unsigned short f2bf(float f) {
    unsigned u = __float_as_uint(f);
    u += 0x7FFFu + ((u >> 16) & 1u);      // round-to-nearest-even
    return (unsigned short)(u >> 16);
}
__device__ __forceinline__ float sigm(float x) { return 1.f / (1.f + __expf(-x)); }

// ---------------------------------------------------------------------------
// Transpose + f32->bf16 convert: src[K][N] (f32) -> dst[Np][K] (bf16), zero pad
// ---------------------------------------------------------------------------
__global__ void k_transpose_bf16(const float* __restrict__ src,
                                 unsigned short* __restrict__ dst,
                                 int K, int N, int Np) {
    __shared__ float tile[32][33];
    int n0 = blockIdx.x * 32, k0 = blockIdx.y * 32;
    int tx = threadIdx.x, ty = threadIdx.y;
    int kk = k0 + ty, nn = n0 + tx;
    tile[ty][tx] = (kk < K && nn < N) ? src[(size_t)kk * N + nn] : 0.f;
    __syncthreads();
    int nd = n0 + ty, kd = k0 + tx;
    if (nd < Np && kd < K) dst[(size_t)nd * K + kd] = f2bf(tile[tx][ty]);
}

// ---------------------------------------------------------------------------
// x = concat(prev_ctx, emb[decoder_input]) @ W_gx + b_gx       [B,E]
// ---------------------------------------------------------------------------
__global__ void k_x(const int* __restrict__ di, const float* __restrict__ pctx,
                    const float* __restrict__ emb, const float* __restrict__ Wgx,
                    const float* __restrict__ bgx, float* __restrict__ xout) {
    int b = blockIdx.x, n = threadIdx.x;                 // blockDim = E (256)
    const float* er = emb + (size_t)di[b] * E;
    const float* pc = pctx + b * D2;
    float acc = bgx[n];
    for (int k = 0; k < KIN; k++) {
        float iv = (k < D2) ? pc[k] : er[k - D2];
        acc += iv * Wgx[(size_t)k * E + n];
    }
    xout[b * E + n] = acc;
}

// ---------------------------------------------------------------------------
// LSTM cell (torch gate order i,f,g,o): h1,c1 written straight into d_out
// ---------------------------------------------------------------------------
__global__ void k_lstm(const float* __restrict__ x, const float* __restrict__ h0,
                       const float* __restrict__ c0, const float* __restrict__ Wih,
                       const float* __restrict__ Whh, const float* __restrict__ bih,
                       const float* __restrict__ bhh,
                       float* __restrict__ h1, float* __restrict__ c1) {
    int b = blockIdx.x;
    int d = blockIdx.y * 256 + threadIdx.x;              // grid (B,2), D=512
    float ai = bih[d]         + bhh[d];
    float af = bih[D + d]     + bhh[D + d];
    float ag = bih[2 * D + d] + bhh[2 * D + d];
    float ao = bih[3 * D + d] + bhh[3 * D + d];
    const float* xr = x + b * E;
    for (int k = 0; k < E; k++) {
        float xv = xr[k];
        const float* w = Wih + (size_t)k * D4;
        ai += xv * w[d]; af += xv * w[D + d]; ag += xv * w[2 * D + d]; ao += xv * w[3 * D + d];
    }
    const float* hr = h0 + b * D;
    for (int k = 0; k < D; k++) {
        float hv = hr[k];
        const float* w = Whh + (size_t)k * D4;
        ai += hv * w[d]; af += hv * w[D + d]; ag += hv * w[2 * D + d]; ao += hv * w[3 * D + d];
    }
    float cc = sigm(af) * c0[b * D + d] + sigm(ai) * FAST_TANH(ag);
    float hh = sigm(ao) * FAST_TANH(cc);
    c1[b * D + d] = cc;
    h1[b * D + d] = hh;
}

// ---------------------------------------------------------------------------
// dec_feat = [h1,c1] @ W_s + b_s         [B,2D]
// ---------------------------------------------------------------------------
__global__ void k_decfeat(const float* __restrict__ h1, const float* __restrict__ c1,
                          const float* __restrict__ Ws, const float* __restrict__ bs,
                          float* __restrict__ df) {
    int b = blockIdx.x;
    int n = blockIdx.y * 256 + threadIdx.x;              // grid (B,4)
    float acc = bs[n];
    const float* hr = h1 + b * D;
    const float* cr = c1 + b * D;
    for (int k = 0; k < D; k++) acc += hr[k] * Ws[(size_t)k * D2 + n];
    for (int k = 0; k < D; k++) acc += cr[k] * Ws[(size_t)(D + k) * D2 + n];
    df[b * D2 + n] = acc;
}

// ---------------------------------------------------------------------------
// Attention scores via WMMA bf16:
//   scores[b,t] = sum_n tanh( (enc@W_h)[b,t,n] + dec_feat[b,n] + cov[b,t]*W_c[n] ) * v[n]
// 1 WG = (b, 16-row t-tile), 4 waves split 64 N-tiles; K accumulated in f32.
// A tile (16x1024) converted to bf16 in LDS once; B from pre-transposed W_hT.
// ---------------------------------------------------------------------------
__global__ void __launch_bounds__(128)
k_attn(const float* __restrict__ enc, const unsigned short* __restrict__ WhT,
       const float* __restrict__ df, const float* __restrict__ cov,
       const float* __restrict__ Wc, const float* __restrict__ vv,
       float* __restrict__ scores) {
    __shared__ __align__(16) unsigned short Albs[16 * D2];   // 32 KB bf16 A tile
    __shared__ float covt[16];
    __shared__ float score_s[16];
    int tt = blockIdx.x, b = blockIdx.y, tid = threadIdx.x;
    int t0 = tt * 16;

    // stage: 16 rows x 1024 cols f32 -> bf16 LDS (row-major)
    const float* erow = enc + ((size_t)b * T + t0) * D2;
    for (int i = 0; i < 32; i++) {
        int f = (tid + i * 128) * 4;                     // 16384 floats total
        fv4 val = *(const fv4*)(erow + f);
        uv2 pk;
        pk[0] = (unsigned)f2bf(val[0]) | ((unsigned)f2bf(val[1]) << 16);
        pk[1] = (unsigned)f2bf(val[2]) | ((unsigned)f2bf(val[3]) << 16);
        *(uv2*)&Albs[f] = pk;
    }
    if (tid < 16)      covt[tid] = cov[b * T + t0 + tid];
    else if (tid < 32) score_s[tid - 16] = 0.f;
    __syncthreads();

    int w = tid >> 5, lane = tid & 31;
    int nl = lane & 15, kh = lane >> 4;
    int m = nl;                                          // A row for this lane
    float covr[8];
    #pragma unroll
    for (int r = 0; r < 8; r++) covr[r] = covt[r + 8 * kh];
    float acc[8] = {0, 0, 0, 0, 0, 0, 0, 0};

    for (int nt = w; nt < 64; nt += 4) {
        int n = nt * 16 + nl;
        v8f c = {0.f, 0.f, 0.f, 0.f, 0.f, 0.f, 0.f, 0.f};
        const unsigned short* bb = WhT + (size_t)n * D2;
        #pragma unroll 4
        for (int k0 = 0; k0 < D2; k0 += 32) {
            FragBF A, Bf;
            const uv4* ap = (const uv4*)&Albs[m * D2 + k0 + kh * 16];
            A.u[0] = ap[0]; A.u[1] = ap[1];              // K = kh*16 .. +15 contiguous
            const uv4* bp = (const uv4*)(bb + k0 + kh * 16);
            Bf.u[0] = bp[0]; Bf.u[1] = bp[1];            // K = kh*16 .. +15 contiguous
            c = __builtin_amdgcn_wmma_f32_16x16x32_bf16(false, A.v, false, Bf.v,
                                                        (short)0, c, false, false);
        }
        float dfe = df[b * D2 + n];
        float wc = Wc[n], vn = vv[n];
        #pragma unroll
        for (int r = 0; r < 8; r++) {                    // row = r + 8*kh, col = n
            float e = FAST_TANH(c[r] + dfe + covr[r] * wc);
            acc[r] += e * vn;
        }
    }
    // reduce over 16 columns inside each lane-half, then across waves via LDS
    #pragma unroll
    for (int r = 0; r < 8; r++) {
        float s = acc[r];
        s += __shfl_xor(s, 1, 32);
        s += __shfl_xor(s, 2, 32);
        s += __shfl_xor(s, 4, 32);
        s += __shfl_xor(s, 8, 32);
        if (nl == 0) atomicAdd(&score_s[r + 8 * kh], s);
    }
    __syncthreads();
    if (tid < 16) scores[b * T + t0 + tid] = score_s[tid];
}

// ---------------------------------------------------------------------------
// att = softmax(scores)*mask renormalized; coverage_next = coverage + att
// ---------------------------------------------------------------------------
__global__ void k_softmax_att(const float* __restrict__ scores, const float* __restrict__ mask,
                              const float* __restrict__ cov, float* __restrict__ att_out,
                              float* __restrict__ cov_out) {
    __shared__ float red[256];
    int b = blockIdx.x, tid = threadIdx.x;
    const float* sr = scores + b * T;
    float sv[4];
    float mx = -1e30f;
    #pragma unroll
    for (int i = 0; i < 4; i++) { sv[i] = sr[tid + i * 256]; mx = fmaxf(mx, sv[i]); }
    red[tid] = mx; __syncthreads();
    for (int s = 128; s > 0; s >>= 1) { if (tid < s) red[tid] = fmaxf(red[tid], red[tid + s]); __syncthreads(); }
    mx = red[0]; __syncthreads();
    float e[4], sum = 0.f;
    #pragma unroll
    for (int i = 0; i < 4; i++) {
        e[i] = __expf(sv[i] - mx) * mask[b * T + tid + i * 256];
        sum += e[i];
    }
    red[tid] = sum; __syncthreads();
    for (int s = 128; s > 0; s >>= 1) { if (tid < s) red[tid] += red[tid + s]; __syncthreads(); }
    float inv = 1.f / red[0];
    #pragma unroll
    for (int i = 0; i < 4; i++) {
        int t = tid + i * 256;
        float a = e[i] * inv;
        att_out[b * T + t] = a;
        cov_out[b * T + t] = cov[b * T + t] + a;
    }
}

// ---------------------------------------------------------------------------
// context[b,d] = sum_t att[b,t] * enc[b,t,d]   (coalesced across d per t)
// ---------------------------------------------------------------------------
__global__ void k_context(const float* __restrict__ att, const float* __restrict__ enc,
                          float* __restrict__ ctx) {
    __shared__ float a_s[T];
    int b = blockIdx.x;
    int d = blockIdx.y * 256 + threadIdx.x;              // grid (B,4)
    for (int i = threadIdx.x; i < T; i += 256) a_s[i] = att[b * T + i];
    __syncthreads();
    const float* eb = enc + (size_t)b * T * D2 + d;
    float acc = 0.f;
    for (int t = 0; t < T; t++) acc += a_s[t] * eb[(size_t)t * D2];
    ctx[b * D2 + d] = acc;
}

// ---------------------------------------------------------------------------
// p_gen = sigmoid([context, h1, c1, x] . W_pg + b_pg)
// ---------------------------------------------------------------------------
__global__ void k_pgen(const float* __restrict__ ctx, const float* __restrict__ h1,
                       const float* __restrict__ c1, const float* __restrict__ x,
                       const float* __restrict__ Wpg, const float* __restrict__ bpg,
                       float* __restrict__ pgen) {
    __shared__ float red[256];
    int b = blockIdx.x, tid = threadIdx.x;
    float acc = 0.f;
    for (int j = tid; j < PGK; j += 256) {
        float xv;
        if (j < D2)            xv = ctx[b * D2 + j];
        else if (j < D2 + D)   xv = h1[b * D + (j - D2)];
        else if (j < 2 * D2)   xv = c1[b * D + (j - D2 - D)];
        else                   xv = x[b * E + (j - 2 * D2)];
        acc += xv * Wpg[j];
    }
    red[tid] = acc; __syncthreads();
    for (int s = 128; s > 0; s >>= 1) { if (tid < s) red[tid] += red[tid + s]; __syncthreads(); }
    if (tid == 0) pgen[b] = sigm(red[0] + bpg[0]);
}

// ---------------------------------------------------------------------------
// hid = [h1, context] @ W_v1 + b_v1      [B,D]
// ---------------------------------------------------------------------------
__global__ void k_hid(const float* __restrict__ h1, const float* __restrict__ ctx,
                      const float* __restrict__ Wv1, const float* __restrict__ bv1,
                      float* __restrict__ hid) {
    int b = blockIdx.x;
    int d = blockIdx.y * 256 + threadIdx.x;              // grid (B,2)
    float acc = bv1[d];
    const float* hr = h1 + b * D;
    const float* cr = ctx + b * D2;
    for (int k = 0; k < D; k++)  acc += hr[k] * Wv1[(size_t)k * D + d];
    for (int k = 0; k < D2; k++) acc += cr[k] * Wv1[(size_t)(D + k) * D + d];
    hid[b * D + d] = acc;
}

// ---------------------------------------------------------------------------
// logits = hid @ W_v2 + b_v2 via WMMA bf16 (M=32 = 2 row tiles, K=512)
// ---------------------------------------------------------------------------
__global__ void __launch_bounds__(128)
k_logits_wmma(const float* __restrict__ hid, const unsigned short* __restrict__ Wv2T,
              const float* __restrict__ bv2, float* __restrict__ logits) {
    __shared__ __align__(16) unsigned short Hb[32 * D];  // 32 KB bf16 A
    int tid = threadIdx.x;
    for (int i = 0; i < 32; i++) {
        int f = (tid + i * 128) * 4;                     // 16384 floats = 32x512
        fv4 val = *(const fv4*)(hid + f);
        uv2 pk;
        pk[0] = (unsigned)f2bf(val[0]) | ((unsigned)f2bf(val[1]) << 16);
        pk[1] = (unsigned)f2bf(val[2]) | ((unsigned)f2bf(val[3]) << 16);
        *(uv2*)&Hb[f] = pk;
    }
    __syncthreads();
    int w = tid >> 5, lane = tid & 31, nl = lane & 15, kh = lane >> 4;
    int nt = blockIdx.x * 4 + w;
    if (nt * 16 >= NPAD) return;
    int n = nt * 16 + nl;
    const unsigned short* bb = Wv2T + (size_t)n * D;
    for (int rt = 0; rt < 2; rt++) {
        v8f c = {0.f, 0.f, 0.f, 0.f, 0.f, 0.f, 0.f, 0.f};
        #pragma unroll
        for (int k0 = 0; k0 < D; k0 += 32) {
            FragBF A, Bf;
            const uv4* ap = (const uv4*)&Hb[(rt * 16 + nl) * D + k0 + kh * 16];
            A.u[0] = ap[0]; A.u[1] = ap[1];
            const uv4* bp = (const uv4*)(bb + k0 + kh * 16);
            Bf.u[0] = bp[0]; Bf.u[1] = bp[1];
            c = __builtin_amdgcn_wmma_f32_16x16x32_bf16(false, A.v, false, Bf.v,
                                                        (short)0, c, false, false);
        }
        if (n < V) {
            float bias = bv2[n];
            #pragma unroll
            for (int r = 0; r < 8; r++) {
                int row = rt * 16 + r + 8 * kh;
                logits[(size_t)row * V + n] = c[r] + bias;
            }
        }
    }
}

// fallback (if workspace too small for bf16 W_v2^T)
__global__ void k_logits_plain(const float* __restrict__ hid, const float* __restrict__ Wv2,
                               const float* __restrict__ bv2, float* __restrict__ logits) {
    int b = blockIdx.x;
    int n = blockIdx.y * 256 + threadIdx.x;
    if (n >= V) return;
    const float* hr = hid + b * D;
    float acc = bv2[n];
    for (int k = 0; k < D; k++) acc += hr[k] * Wv2[(size_t)k * V + n];
    logits[(size_t)b * V + n] = acc;
}

// ---------------------------------------------------------------------------
// vocab softmax, scale by p_gen, write full [B, V+EXTRA] rows (extras = 0)
// ---------------------------------------------------------------------------
__global__ void k_vocab(const float* __restrict__ logits, const float* __restrict__ pgen,
                        float* __restrict__ out) {
    __shared__ float red[256];
    int b = blockIdx.x, tid = threadIdx.x;
    const float* lg = logits + (size_t)b * V;
    float* orow = out + (size_t)b * VE;
    float mx = -1e30f;
    for (int n = tid; n < V; n += 256) mx = fmaxf(mx, lg[n]);
    red[tid] = mx; __syncthreads();
    for (int s = 128; s > 0; s >>= 1) { if (tid < s) red[tid] = fmaxf(red[tid], red[tid + s]); __syncthreads(); }
    mx = red[0]; __syncthreads();
    float sum = 0.f;
    for (int n = tid; n < V; n += 256) sum += __expf(lg[n] - mx);
    red[tid] = sum; __syncthreads();
    for (int s = 128; s > 0; s >>= 1) { if (tid < s) red[tid] += red[tid + s]; __syncthreads(); }
    float scale = pgen[b] / red[0];
    for (int n = tid; n < V; n += 256)       orow[n] = scale * __expf(lg[n] - mx);
    for (int n = V + tid; n < VE; n += 256)  orow[n] = 0.f;
}

// ---------------------------------------------------------------------------
// scatter-add (1-p_gen)*att into extended-vocab distribution
// ---------------------------------------------------------------------------
__global__ void k_scatter(const int* __restrict__ eiv, const float* __restrict__ att,
                          const float* __restrict__ pgen, float* __restrict__ out) {
    int b = blockIdx.x;
    int t = blockIdx.y * 256 + threadIdx.x;              // grid (B,4)
    float om = 1.f - pgen[b];
    int idx = eiv[b * T + t];
    atomicAdd(out + (size_t)b * VE + idx, om * att[b * T + t]);
}

// ---------------------------------------------------------------------------
extern "C" void kernel_launch(void* const* d_in, const int* in_sizes, int n_in,
                              void* d_out, int out_size, void* d_ws, size_t ws_size,
                              hipStream_t stream) {
    (void)in_sizes; (void)n_in; (void)out_size;
    const int*   dec_in = (const int*)d_in[0];
    const float* h0   = (const float*)d_in[1];
    const float* c0   = (const float*)d_in[2];
    const float* enc  = (const float*)d_in[3];
    const float* mask = (const float*)d_in[4];
    const float* pctx = (const float*)d_in[5];
    const float* cov  = (const float*)d_in[6];
    const int*   eiv  = (const int*)d_in[8];
    const float* emb  = (const float*)d_in[10];
    const float* Wgx  = (const float*)d_in[11];
    const float* bgx  = (const float*)d_in[12];
    const float* Wih  = (const float*)d_in[13];
    const float* Whh  = (const float*)d_in[14];
    const float* bih  = (const float*)d_in[15];
    const float* bhh  = (const float*)d_in[16];
    const float* Wh   = (const float*)d_in[17];
    const float* Wc   = (const float*)d_in[18];
    const float* Ws   = (const float*)d_in[19];
    const float* bs   = (const float*)d_in[20];
    const float* vv   = (const float*)d_in[21];
    const float* Wpg  = (const float*)d_in[22];
    const float* bpg  = (const float*)d_in[23];
    const float* Wv1  = (const float*)d_in[24];
    const float* bv1  = (const float*)d_in[25];
    const float* Wv2  = (const float*)d_in[26];
    const float* bv2  = (const float*)d_in[27];

    float* out   = (float*)d_out;
    float* o_h1  = out + (size_t)B * VE;
    float* o_c1  = o_h1 + B * D;
    float* o_ctx = o_c1 + B * D;
    float* o_att = o_ctx + B * D2;
    float* o_cov = o_att + B * T;

    // workspace layout (16B-aligned subregions)
    float* W      = (float*)d_ws;
    float* ws_x   = W;                       // B*E     = 8192
    float* ws_df  = ws_x   + B * E;          // B*D2    = 32768
    float* ws_sc  = ws_df  + B * D2;         // B*T     = 32768
    float* ws_hid = ws_sc  + B * T;          // B*D     = 16384
    float* ws_pg  = ws_hid + B * D;          // 32
    float* ws_log = ws_pg  + 32;             // B*V
    float* fend   = ws_log + (size_t)B * V;
    unsigned short* WhT  = (unsigned short*)fend;                     // 1024*1024 bf16
    unsigned short* Wv2T = (unsigned short*)(fend + ((size_t)D2 * D2) / 2);  // NPAD*512 bf16
    size_t need_full = (size_t)((char*)(Wv2T + (size_t)NPAD * D) - (char*)d_ws);
    bool big = ws_size >= need_full;

    dim3 b32(32, 32);
    k_transpose_bf16<<<dim3(D2 / 32, D2 / 32), b32, 0, stream>>>(Wh, WhT, D2, D2, D2);
    if (big)
        k_transpose_bf16<<<dim3((NPAD + 31) / 32, D / 32), b32, 0, stream>>>(Wv2, Wv2T, D, V, NPAD);

    k_x      <<<dim3(B),    dim3(E),   0, stream>>>(dec_in, pctx, emb, Wgx, bgx, ws_x);
    k_lstm   <<<dim3(B, 2), dim3(256), 0, stream>>>(ws_x, h0, c0, Wih, Whh, bih, bhh, o_h1, o_c1);
    k_decfeat<<<dim3(B, 4), dim3(256), 0, stream>>>(o_h1, o_c1, Ws, bs, ws_df);
    k_attn   <<<dim3(T / 16, B), dim3(128), 0, stream>>>(enc, WhT, ws_df, cov, Wc, vv, ws_sc);
    k_softmax_att<<<dim3(B), dim3(256), 0, stream>>>(ws_sc, mask, cov, o_att, o_cov);
    k_context<<<dim3(B, 4), dim3(256), 0, stream>>>(o_att, enc, o_ctx);
    k_pgen   <<<dim3(B),    dim3(256), 0, stream>>>(o_ctx, o_h1, o_c1, ws_x, Wpg, bpg, ws_pg);
    k_hid    <<<dim3(B, 2), dim3(256), 0, stream>>>(o_h1, o_ctx, Wv1, bv1, ws_hid);
    if (big)
        k_logits_wmma<<<dim3((NPAD / 16 + 3) / 4), dim3(128), 0, stream>>>(ws_hid, Wv2T, bv2, ws_log);
    else
        k_logits_plain<<<dim3(B, (V + 255) / 256), dim3(256), 0, stream>>>(ws_hid, Wv2, bv2, ws_log);
    k_vocab  <<<dim3(B),    dim3(256), 0, stream>>>(ws_log, ws_pg, out);
    k_scatter<<<dim3(B, T / 256), dim3(256), 0, stream>>>(eiv, o_att, ws_pg, out);
}
